// PointerDecoder_31473520345559
// MI455X (gfx1250) — matrix operand order repaired and checked by reading, not compiled, not run on hardware
//
#include <hip/hip_runtime.h>
#include <cstdint>
#include <cstddef>

#define DEVFN __device__ __forceinline__

typedef __bf16 bf16_t;
typedef bf16_t v16bf __attribute__((ext_vector_type(16)));
typedef float  v8f   __attribute__((ext_vector_type(8)));

// ---------------- problem dims ----------------
static constexpr int B = 64, L = 512, E = 128, H = 256;
static constexpr int T = 512;          // decode steps == L

// ---------------- workspace layout (bytes) ----------------
static constexpr size_t OFF_CTX   = 0;                                   // B*L*H f32 (32 MB)
static constexpr size_t OFF_WCAT  = OFF_CTX   + (size_t)B*L*H*4;         // 64nt*12kt*512 bf16
static constexpr size_t OFF_WQ    = OFF_WCAT  + 393216ull*2;             // 16nt*8kt*512 bf16
static constexpr size_t OFF_WO    = OFF_WQ    + 65536ull*2;              // 16nt*16kt*512 bf16
static constexpr size_t OFF_WCP   = OFF_WO    + 131072ull*2;             // 16nt*8kt*512 bf16 (Wc)
static constexpr size_t OFF_X     = OFF_WCP   + 65536ull*2;              // B*E f32
static constexpr size_t OFF_HPREV = OFF_X     + (size_t)B*E*4;           // B*H f32
static constexpr size_t OFF_HT    = OFF_HPREV + (size_t)B*H*4;           // B*H f32
static constexpr size_t OFF_C     = OFF_HT    + (size_t)B*H*4;           // B*H f32
static constexpr size_t OFF_MASK  = OFF_C     + (size_t)B*H*4;           // B*L f32
static constexpr size_t OFF_Q     = OFF_MASK  + (size_t)B*L*4;           // B*H f32
static constexpr size_t OFF_CONC  = OFF_Q     + (size_t)B*H*4;           // B*2H f32
static constexpr size_t OFF_BAR   = OFF_CONC  + (size_t)B*2*H*4;         // u32
static constexpr size_t WS_NEED   = OFF_BAR + 256;

// ---------------- output layout (f32 elements) ----------------
static constexpr size_t OUT_PTR = (size_t)B*L*T;        // alphas (B,L,T) first
static constexpr size_t OUT_HF  = OUT_PTR + (size_t)B*T;
static constexpr size_t OUT_CF  = OUT_HF  + (size_t)B*H;

// ---------------- small math helpers ----------------
DEVFN float fsig(float x)   { return 1.0f / (1.0f + __expf(-x)); }
DEVFN float ftanh_(float x) { float e = __expf(2.0f*x); return 1.0f - 2.0f/(e + 1.0f); }
DEVFN unsigned short f2bf(float f) {
  union { bf16_t h; unsigned short u; } v; v.h = (bf16_t)f; return v.u;
}

// ---------------- WMMA fragment helpers (gfx1250 bf16 16x16x32) ----------------
// A fragment (16x32 bf16), ISA 7.12.2: lane%16 = row M,
// VGPR v<4 -> K = 2v,2v+1 (+8 for lanes>=16); VGPR v>=4 -> K = 16+2(v-4).. (+8)
template <typename G>
DEVFN v16bf make_afrag(int lane, G&& g) {
  v16bf a;
  const int koff = (lane & 16) ? 8 : 0;
#pragma unroll
  for (int e = 0; e < 16; ++e) {
    const int vv = e >> 1;
    const int kb = (vv < 4) ? (vv * 2) : (16 + (vv - 4) * 2);
    a[e] = (bf16_t)g(kb + koff + (e & 1));
  }
  return a;
}

// B fragment (32x16 bf16): lane%16 = col N; element e -> K = 8h+e (e<8),
// 16+8h+(e-8) (e>=8), h = lane/16. Pre-packed: 16 contiguous bf16 per lane.
DEVFN v16bf load_bfrag(const unsigned short* Bp, int block, int lane) {
  return *(const v16bf*)(Bp + (size_t)block*512 + (size_t)lane*16);
}

DEVFN v8f wmma_bf16(v16bf a, v16bf b, v8f c) {
  return __builtin_amdgcn_wmma_f32_16x16x32_bf16(false, a, false, b, (short)0, c,
                                                 false, false);
}

// Partial-K GEMM for one 16x16 tile with two interleaved accumulators so
// back-to-back WMMAs are independent (avoids the D->C RAW hazard chain).
template <typename G>
DEVFN v8f gemm_partial(const unsigned short* __restrict__ Bp, int blkbase,
                       int kt0, int kt1, int lane, G&& srcA) {
  v8f a0 = {}, a1 = {};
  for (int kt = kt0; kt < kt1; ++kt) {
    v16bf a = make_afrag(lane, [&](int kl) { return srcA(kt*32 + kl); });
    v16bf b = load_bfrag(Bp, blkbase + kt, lane);
    if ((kt - kt0) & 1) a1 = wmma_bf16(a, b, a1);
    else                a0 = wmma_bf16(a, b, a0);
  }
  return a0 + a1;
}

// ---------------- device-wide barrier (monotonic epoch counter) ----------------
DEVFN void grid_barrier(unsigned* bar, unsigned& epoch) {
  __threadfence();          // flush this thread's global stores
  __syncthreads();
  if (threadIdx.x == 0) {
    atomicAdd(bar, 1u);
    epoch += (unsigned)B;
    while (__hip_atomic_load(bar, __ATOMIC_ACQUIRE, __HIP_MEMORY_SCOPE_AGENT) < epoch)
      __builtin_amdgcn_s_sleep(1);
  }
  __syncthreads();
}

// ================= kernel 1: init recurrent state + barrier =================
__global__ __launch_bounds__(256) void init_state(
    const float* __restrict__ dec, const float* __restrict__ h0,
    const float* __restrict__ c0, float* __restrict__ xg,
    float* __restrict__ hprev, float* __restrict__ cg,
    float* __restrict__ maskg, unsigned* __restrict__ bar) {
  const int i = blockIdx.x*256 + threadIdx.x;
  if (i == 0) *bar = 0u;
  if (i < B*E) xg[i] = dec[i];
  if (i < B*H) { hprev[i] = h0[i]; cg[i] = c0[i]; }
  if (i < B*L) maskg[i] = 1.0f;
}

// ================= kernel 2: pack weights into bf16 B-fragment blocks =========
__global__ __launch_bounds__(256) void pack_weights(
    const float* __restrict__ Wih, const float* __restrict__ Whh,
    const float* __restrict__ Wq,  const float* __restrict__ Wo,
    const float* __restrict__ Wc,
    unsigned short* __restrict__ WcatP, unsigned short* __restrict__ WqP,
    unsigned short* __restrict__ WoP,   unsigned short* __restrict__ WcP) {
  const int i = blockIdx.x*256 + threadIdx.x;
  const int r = i & 511, lane = r >> 4, e = r & 15;
  const int klane = ((lane & 16) ? 8 : 0) + ((e < 8) ? e : (e + 8));
  const int nlane = lane & 15;
  if (i < 393216) {                       // Wcat: 64 nt x 12 kt, N=1024 K=384
    const int blk = i >> 9, nt = blk / 12, kt = blk % 12;
    const int col = nt*16 + nlane, k = kt*32 + klane;
    const float w = (k < E) ? Wih[(size_t)k*1024 + col]
                            : Whh[(size_t)(k - E)*1024 + col];
    WcatP[i] = f2bf(w);
  } else if (i < 458752) {                // Wq: 16 x 8, N=256 K=256
    const int j = i - 393216, blk = j >> 9, nt = blk / 8, kt = blk % 8;
    WqP[j] = f2bf(Wq[(size_t)(kt*32 + klane)*H + nt*16 + nlane]);
  } else if (i < 589824) {                // Wo: 16 x 16, N=256 K=512
    const int j = i - 458752, blk = j >> 9, nt = blk / 16, kt = blk % 16;
    WoP[j] = f2bf(Wo[(size_t)(kt*32 + klane)*H + nt*16 + nlane]);
  } else if (i < 655360) {                // Wc: 16 x 8, N=256 K=256
    const int j = i - 589824, blk = j >> 9, nt = blk / 8, kt = blk % 8;
    WcP[j] = f2bf(Wc[(size_t)(kt*32 + klane)*H + nt*16 + nlane]);
  }
}

// ================= kernel 3: hoisted ctx' = context @ Wc + bc (WMMA) =========
__global__ __launch_bounds__(256) void ctx_gemm(
    const float* __restrict__ context, const unsigned short* __restrict__ WcP,
    const float* __restrict__ bc, float* __restrict__ ctxp) {
  const int wave = threadIdx.x >> 5, lane = threadIdx.x & 31;
  const int tile = blockIdx.x*8 + wave;            // 32768 tiles: M=B*L, N=H
  const int mt = tile >> 4, nt = tile & 15;
  const int row = mt*16 + (lane & 15);
  const int col = nt*16 + (lane & 15);
  v8f acc = gemm_partial(WcP, nt*8, 0, 8, lane, [&](int k) {
    return context[(size_t)row*H + k];
  });
  const int rbase = mt*16 + ((lane & 16) ? 8 : 0);
#pragma unroll
  for (int r = 0; r < 8; ++r)
    ctxp[(size_t)(rbase + r)*H + col] = acc[r] + bc[col];
}

// ================= kernel 4: persistent decoder (64 WGs x 256) ===============
__global__ __launch_bounds__(256) void decoder_persist(
    const float* __restrict__ embed, const float* __restrict__ bih,
    const float* __restrict__ bhh,  const float* __restrict__ bo,
    const float* __restrict__ bq,   const float* __restrict__ Vvec,
    const float* __restrict__ ctxp,
    const unsigned short* __restrict__ WcatP,
    const unsigned short* __restrict__ WqP,
    const unsigned short* __restrict__ WoP,
    float* __restrict__ xg, float* __restrict__ hprev,
    float* __restrict__ htg, float* __restrict__ cg,
    float* __restrict__ maskg, float* __restrict__ qg,
    float* __restrict__ conc, unsigned* __restrict__ bar,
    float* __restrict__ out) {
  const int tid = threadIdx.x, wg = blockIdx.x;
  const int wave = tid >> 5, lane = tid & 31;
  unsigned epoch = 0;

  __shared__ float gsm[8][16][16];   // gate quadrant x K-half partial tiles
  __shared__ float qs[H], Vs[H];
  __shared__ float atts[L];
  __shared__ float accs[8 * H];      // 8 partial 16x16 tiles / flash accumulators
  __shared__ float ms[8], ss[8];
  __shared__ float rv[256];
  __shared__ int   ri[256];

  Vs[tid] = Vvec[tid];
  __syncthreads();

  const int mg = wg >> 4;      // 0..3 : 16-row batch tile owned in phase A
  const int js = wg & 15;      // 0..15: 16-wide H slice owned in phase A
  const int mt4 = wave & 3;    // m-tile for split-K phases B/D
  const int kh  = wave >> 2;   // K-half

  for (int t = 0; t < T; ++t) {
    // ---- Phase A: gates GEMM (8 waves: quadrant x K-half) + fused LSTM ----
    {
      const int qd = wave & 3;                      // gate quadrant (i,f,g,o)
      const int ntile = qd*16 + js;                 // N-tile within 4H=1024
      const int m = mg*16 + (lane & 15);
      v8f p = gemm_partial(WcatP, ntile*12, kh*6, kh*6 + 6, lane, [&](int k) {
        return (k < E) ? xg[m*E + k] : hprev[m*H + (k - E)];
      });
      const int cl = lane & 15, rb = (lane & 16) ? 8 : 0;
#pragma unroll
      for (int r = 0; r < 8; ++r) gsm[wave][rb + r][cl] = p[r];
    }
    __syncthreads();
    {
      const int rl = tid >> 4, jl = tid & 15;
      const int b = mg*16 + rl, j = js*16 + jl;
      const float ig = gsm[0][rl][jl] + gsm[4][rl][jl] + bih[j]       + bhh[j];
      const float fg = gsm[1][rl][jl] + gsm[5][rl][jl] + bih[H + j]   + bhh[H + j];
      const float gg = gsm[2][rl][jl] + gsm[6][rl][jl] + bih[2*H + j] + bhh[2*H + j];
      const float og = gsm[3][rl][jl] + gsm[7][rl][jl] + bih[3*H + j] + bhh[3*H + j];
      const float ct = fsig(fg)*cg[b*H + j] + fsig(ig)*ftanh_(gg);
      cg[b*H + j]  = ct;
      htg[b*H + j] = fsig(og)*ftanh_(ct);
    }
    grid_barrier(bar, epoch);

    // ---- Phase B: q = h_t @ Wq + bq (WGs 0..15; 8 waves, split-K) ----
    if (wg < 16) {
      const int m = mt4*16 + (lane & 15);
      v8f p = gemm_partial(WqP, wg*8, kh*4, kh*4 + 4, lane, [&](int k) {
        return htg[m*H + k];
      });
      const int cl = lane & 15, rb = (lane & 16) ? 8 : 0;
#pragma unroll
      for (int r = 0; r < 8; ++r) accs[wave*256 + (rb + r)*16 + cl] = p[r];
      __syncthreads();
      const int coll = tid & 15, rowl = tid >> 4;
      const int col = wg*16 + coll;
#pragma unroll
      for (int m2 = 0; m2 < 4; ++m2) {
        const float v = accs[m2*256 + tid] + accs[(m2 + 4)*256 + tid];
        qg[(m2*16 + rowl)*H + col] = v + bq[col];
      }
    }
    grid_barrier(bar, epoch);

    // ---- Phase C: additive attention, one pass over ctx' (flash softmax) ----
    {
      const int b = wg;
      qs[tid] = qg[b*H + tid];
      __syncthreads();
      const int hb = lane*8;                // lane owns 8 consecutive H elems
      float mrun = -3.0e38f, srun = 0.0f;
      float acc[8];
#pragma unroll
      for (int u = 0; u < 8; ++u) acc[u] = 0.0f;
      for (int li = 0; li < 64; ++li) {     // wave owns contiguous 64 positions
        const int l = wave*64 + li;
        const float* cp = ctxp + ((size_t)(b*L + l))*H + hb;
        if (li + 1 < 64) __builtin_prefetch(cp + H, 0, 1);
        const float4* cp4 = (const float4*)cp;
        const float4 cA = cp4[0], cB = cp4[1];
        const float cv[8] = {cA.x, cA.y, cA.z, cA.w, cB.x, cB.y, cB.z, cB.w};
        float v = 0.0f;
#pragma unroll
        for (int u = 0; u < 8; ++u)
          v += Vs[hb + u] * ftanh_(qs[hb + u] + cv[u]);
#pragma unroll
        for (int off = 16; off >= 1; off >>= 1) v += __shfl_xor(v, off, 32);
        const float att = (maskg[b*L + l] == 0.0f) ? -1.0e9f : v;
        if (lane == 0) atts[l] = att;
        const float mnew = fmaxf(mrun, att);
        const float sc = __expf(mrun - mnew);
        const float p  = __expf(att - mnew);
        srun = srun*sc + p;
#pragma unroll
        for (int u = 0; u < 8; ++u) acc[u] = acc[u]*sc + p*cv[u];
        mrun = mnew;
      }
      if (lane == 0) { ms[wave] = mrun; ss[wave] = srun; }
#pragma unroll
      for (int u = 0; u < 8; ++u) accs[wave*H + hb + u] = acc[u];
      __syncthreads();
      float mstar = ms[0];
#pragma unroll
      for (int w = 1; w < 8; ++w) mstar = fmaxf(mstar, ms[w]);
      float sstar = 0.0f;
#pragma unroll
      for (int w = 0; w < 8; ++w) sstar += ss[w] * __expf(ms[w] - mstar);
      float ah = 0.0f;
#pragma unroll
      for (int w = 0; w < 8; ++w) ah += accs[w*H + tid] * __expf(ms[w] - mstar);
      const float inv = 1.0f / sstar;
      ah *= inv;
      conc[b*(2*H) + tid]     = ah;              // [attn_h | h_t] for Wo GEMM
      conc[b*(2*H) + H + tid] = htg[b*H + tid];
      // alphas out + masked argmax candidates (first-index tie-break)
      float bv = -1.0f; int bi = 0;
#pragma unroll
      for (int r2 = 0; r2 < 2; ++r2) {
        const int l = tid + r2*256;
        const float alpha = __expf(atts[l] - mstar) * inv;
        out[((size_t)(b*L + l))*T + t] = alpha;
        const float mv = alpha * maskg[b*L + l];
        if (mv > bv) { bv = mv; bi = l; }
      }
      rv[tid] = bv; ri[tid] = bi;
      __syncthreads();
      for (int s = 128; s > 0; s >>= 1) {
        if (tid < s) {
          if (rv[tid + s] > rv[tid] ||
              (rv[tid + s] == rv[tid] && ri[tid + s] < ri[tid])) {
            rv[tid] = rv[tid + s]; ri[tid] = ri[tid + s];
          }
        }
        __syncthreads();
      }
      const int idx = ri[0];
      if (tid == 0) {
        maskg[b*L + idx] = 0.0f;
        out[OUT_PTR + (size_t)b*T + t] = (float)idx;
      }
      if (tid < E) xg[b*E + tid] = embed[((size_t)(b*L + idx))*E + tid];
    }
    grid_barrier(bar, epoch);

    // ---- Phase D: h = tanh([attn_h|h_t] @ Wo + bo) (WGs 0..15, split-K) ----
    if (wg < 16) {
      const int m = mt4*16 + (lane & 15);
      v8f p = gemm_partial(WoP, wg*16, kh*8, kh*8 + 8, lane, [&](int k) {
        return conc[m*(2*H) + k];
      });
      const int cl = lane & 15, rb = (lane & 16) ? 8 : 0;
#pragma unroll
      for (int r = 0; r < 8; ++r) accs[wave*256 + (rb + r)*16 + cl] = p[r];
      __syncthreads();
      const int coll = tid & 15, rowl = tid >> 4;
      const int col = wg*16 + coll;
#pragma unroll
      for (int m2 = 0; m2 < 4; ++m2) {
        const float v = accs[m2*256 + tid] + accs[(m2 + 4)*256 + tid];
        hprev[(m2*16 + rowl)*H + col] = ftanh_(v + bo[col]);
      }
    }
    grid_barrier(bar, epoch);
  }

  // final carries
  out[OUT_HF + (size_t)wg*H + tid] = hprev[wg*H + tid];
  out[OUT_CF + (size_t)wg*H + tid] = cg[wg*H + tid];
}

// ================= host entry =================
extern "C" void kernel_launch(void* const* d_in, const int* in_sizes, int n_in,
                              void* d_out, int out_size, void* d_ws, size_t ws_size,
                              hipStream_t stream) {
  (void)in_sizes; (void)n_in; (void)out_size;
  if (ws_size < WS_NEED) return;

  const float* embed = (const float*)d_in[0];
  const float* dec   = (const float*)d_in[1];
  const float* h0    = (const float*)d_in[2];
  const float* c0    = (const float*)d_in[3];
  const float* ctx   = (const float*)d_in[4];
  const float* Wih   = (const float*)d_in[5];
  const float* bih   = (const float*)d_in[6];
  const float* Whh   = (const float*)d_in[7];
  const float* bhh   = (const float*)d_in[8];
  const float* Wo    = (const float*)d_in[9];
  const float* bo    = (const float*)d_in[10];
  const float* Wq    = (const float*)d_in[11];
  const float* bq    = (const float*)d_in[12];
  const float* Wc    = (const float*)d_in[13];
  const float* bc    = (const float*)d_in[14];
  const float* V     = (const float*)d_in[15];

  char* ws = (char*)d_ws;
  float*          ctxp  = (float*)(ws + OFF_CTX);
  unsigned short* WcatP = (unsigned short*)(ws + OFF_WCAT);
  unsigned short* WqP   = (unsigned short*)(ws + OFF_WQ);
  unsigned short* WoP   = (unsigned short*)(ws + OFF_WO);
  unsigned short* WcP   = (unsigned short*)(ws + OFF_WCP);
  float*          xg    = (float*)(ws + OFF_X);
  float*          hprev = (float*)(ws + OFF_HPREV);
  float*          htg   = (float*)(ws + OFF_HT);
  float*          cg    = (float*)(ws + OFF_C);
  float*          maskg = (float*)(ws + OFF_MASK);
  float*          qg    = (float*)(ws + OFF_Q);
  float*          conc  = (float*)(ws + OFF_CONC);
  unsigned*       bar   = (unsigned*)(ws + OFF_BAR);

  init_state  <<<128,  256, 0, stream>>>(dec, h0, c0, xg, hprev, cg, maskg, bar);
  pack_weights<<<2560, 256, 0, stream>>>(Wih, Whh, Wq, Wo, Wc, WcatP, WqP, WoP, WcP);
  ctx_gemm    <<<4096, 256, 0, stream>>>(ctx, WcP, bc, ctxp);
  decoder_persist<<<64, 256, 0, stream>>>(embed, bih, bhh, bo, bq, V, ctxp,
                                          WcatP, WqP, WoP, xg, hprev, htg, cg,
                                          maskg, qg, conc, bar, (float*)d_out);
}
